// SimGCLModel_41523743817917
// MI455X (gfx1250) — compile-verified, matrix operand order checked by reading, not run on hardware
//
#include <hip/hip_runtime.h>
#include <hip/hip_bf16.h>

typedef __attribute__((ext_vector_type(16))) _Float16 v16h;
typedef __attribute__((ext_vector_type(8)))  float    v8f;

#define EPS_C 0.1f

// ---------------------------------------------------------------------------
// init: x_cur = concat(Gu, Gi); zero x_next (atomic accumulator), out (mean
// accumulator) and deg/dinv. Everything we later read is initialized here so
// the 0xAA poison never leaks, and replays are deterministic.
// ---------------------------------------------------------------------------
__global__ void simgcl_init(const float* __restrict__ Gu, const float* __restrict__ Gi,
                            float* __restrict__ x_cur, float* __restrict__ x_next,
                            float* __restrict__ out, float* __restrict__ dinv,
                            int nu_elems, int total_elems, int N) {
  int i = blockIdx.x * blockDim.x + threadIdx.x;
  if (i < total_elems) {
    float v = (i < nu_elems) ? Gu[i] : Gi[i - nu_elems];
    x_cur[i]  = v;
    x_next[i] = 0.0f;
    out[i]    = 0.0f;
  }
  if (i < N) dinv[i] = 0.0f;
}

// deg[dst] += 1 over all edges (stored as float; counts are exact in f32)
__global__ void simgcl_deg(const int* __restrict__ ei, float* __restrict__ deg, int E) {
  int e = blockIdx.x * blockDim.x + threadIdx.x;
  if (e < E) unsafeAtomicAdd(&deg[ei[E + e]], 1.0f);
}

// dinv = deg > 0 ? 1/sqrt(deg) : 0   (in place)
__global__ void simgcl_dinv(float* __restrict__ dinv, int N) {
  int i = blockIdx.x * blockDim.x + threadIdx.x;
  if (i < N) {
    float d = dinv[i];
    dinv[i] = (d > 0.0f) ? (1.0f / sqrtf(d)) : 0.0f;
  }
}

// norm[e] = dinv[src] * dinv[dst]
__global__ void simgcl_norm(const int* __restrict__ ei, const float* __restrict__ dinv,
                            float* __restrict__ norm, int E) {
  int e = blockIdx.x * blockDim.x + threadIdx.x;
  if (e < E) norm[e] = dinv[ei[e]] * dinv[ei[E + e]];
}

// ---------------------------------------------------------------------------
// SpMM: one wave32 per edge; lane handles a float2 column pair. x (57.6 MB)
// and the accumulator (57.6 MB) both live in the 192 MB L2, so the random row
// gather (256B/wave, one b64 per lane) and the native global_atomic_add_f32
// scatter run at L2 speed. Prefetch the edge stream (global_prefetch_b8).
// ---------------------------------------------------------------------------
__global__ void simgcl_spmm(const int* __restrict__ ei, const float* __restrict__ norm,
                            const float* __restrict__ x_cur, float* __restrict__ x_next,
                            int E) {
  unsigned t = blockIdx.x * blockDim.x + threadIdx.x;
  int e    = (int)(t >> 5);
  int lane = (int)(t & 31);
  if (e >= E) return;
  if (lane == 0) {
    int pe = e + 8192;
    if (pe < E) {
      __builtin_prefetch(&ei[pe], 0, 2);
      __builtin_prefetch(&ei[E + pe], 0, 2);
      __builtin_prefetch(&norm[pe], 0, 2);
    }
  }
  int s = ei[e];
  int d = ei[E + e];
  float w = norm[e];
  const float2* xs = (const float2*)(x_cur + (size_t)s * 64);
  float*        xd = x_next + (size_t)d * 64;
  float2 v = xs[lane];
  unsafeAtomicAdd(&xd[lane * 2],     w * v.x);
  unsafeAtomicAdd(&xd[lane * 2 + 1], w * v.y);
}

// ---------------------------------------------------------------------------
// Noise perturbation. Per wave: 16 rows. Row L2-norms computed on the matrix
// core: C(16x16,f32) += A(16x32,f16 of squared noise) x B(ones) twice (K=64),
// so every column of C holds the row sum of squares. A-operand layout follows
// the CDNA5 16-bit 16x32 spec: lane = row | group, elems 0..7 = K kb..kb+7,
// elems 8..15 = K 16+kb..16+kb+7 (kb = (lane>>4)*8); second WMMA covers cols
// 32..63. C layout: VGPR r, lanes 0-15 -> row r, lanes 16-31 -> row r+8.
// Then (float4-vectorized): res = x + sign(x)*noise*(EPS/max(||row||,1e-12));
// write next-layer input, accumulate the layer mean, re-zero the accumulator.
// ---------------------------------------------------------------------------
__global__ void simgcl_noise_wmma(const float* __restrict__ noise_k,
                                  float* __restrict__ x_next,  // SpMM result; zeroed after read
                                  float* __restrict__ x_cur,   // perturbed output
                                  float* __restrict__ out,     // mean accumulator
                                  int N, float inv_layers) {
  __shared__ float scales[8][16];
  int wave = threadIdx.x >> 5;
  int lane = threadIdx.x & 31;
  int r0    = (blockIdx.x * 8 + wave) * 16;
  int row_a = r0 + (lane & 15);
  int kb    = (lane >> 4) * 8;      // 0 or 8
  bool valid = (row_a < N);
  const float4* nz4 = (const float4*)(noise_k + (size_t)(valid ? row_a : 0) * 64);
  int k4 = kb >> 2;                 // float4 index: 0 or 2

  float4 p0 = nz4[k4],      p1 = nz4[k4 + 1];   // cols kb..kb+7
  float4 p2 = nz4[k4 + 4],  p3 = nz4[k4 + 5];   // cols 16+kb..16+kb+7
  float4 p4 = nz4[k4 + 8],  p5 = nz4[k4 + 9];   // cols 32+kb..
  float4 p6 = nz4[k4 + 12], p7 = nz4[k4 + 13];  // cols 48+kb..
  if (!valid) {
    p0 = p1 = p2 = p3 = p4 = p5 = p6 = p7 = make_float4(0.f, 0.f, 0.f, 0.f);
  }

  v16h a0, a1, ones;
  a0[0]  = (_Float16)(p0.x * p0.x); a0[1]  = (_Float16)(p0.y * p0.y);
  a0[2]  = (_Float16)(p0.z * p0.z); a0[3]  = (_Float16)(p0.w * p0.w);
  a0[4]  = (_Float16)(p1.x * p1.x); a0[5]  = (_Float16)(p1.y * p1.y);
  a0[6]  = (_Float16)(p1.z * p1.z); a0[7]  = (_Float16)(p1.w * p1.w);
  a0[8]  = (_Float16)(p2.x * p2.x); a0[9]  = (_Float16)(p2.y * p2.y);
  a0[10] = (_Float16)(p2.z * p2.z); a0[11] = (_Float16)(p2.w * p2.w);
  a0[12] = (_Float16)(p3.x * p3.x); a0[13] = (_Float16)(p3.y * p3.y);
  a0[14] = (_Float16)(p3.z * p3.z); a0[15] = (_Float16)(p3.w * p3.w);
  a1[0]  = (_Float16)(p4.x * p4.x); a1[1]  = (_Float16)(p4.y * p4.y);
  a1[2]  = (_Float16)(p4.z * p4.z); a1[3]  = (_Float16)(p4.w * p4.w);
  a1[4]  = (_Float16)(p5.x * p5.x); a1[5]  = (_Float16)(p5.y * p5.y);
  a1[6]  = (_Float16)(p5.z * p5.z); a1[7]  = (_Float16)(p5.w * p5.w);
  a1[8]  = (_Float16)(p6.x * p6.x); a1[9]  = (_Float16)(p6.y * p6.y);
  a1[10] = (_Float16)(p6.z * p6.z); a1[11] = (_Float16)(p6.w * p6.w);
  a1[12] = (_Float16)(p7.x * p7.x); a1[13] = (_Float16)(p7.y * p7.y);
  a1[14] = (_Float16)(p7.z * p7.z); a1[15] = (_Float16)(p7.w * p7.w);
  for (int j = 0; j < 16; ++j) ones[j] = (_Float16)1.0f;

  v8f c = {};
  c = __builtin_amdgcn_wmma_f32_16x16x32_f16(false, a0, false, ones, (short)0, c, false, false);
  c = __builtin_amdgcn_wmma_f32_16x16x32_f16(false, a1, false, ones, (short)0, c, false, false);

  if (lane == 0)  for (int r = 0; r < 8; ++r) scales[wave][r]     = c[r];
  if (lane == 16) for (int r = 0; r < 8; ++r) scales[wave][r + 8] = c[r];
  __syncthreads();

  // Elementwise pass, float4-vectorized: wave owns 256 float4s (16 rows).
  const float4* nz4all = (const float4*)noise_k;
  float4* xn4 = (float4*)x_next;
  float4* xc4 = (float4*)x_cur;
  float4* ot4 = (float4*)out;
  size_t base4 = (size_t)r0 * 16;
  const float4 zero4 = make_float4(0.f, 0.f, 0.f, 0.f);
  for (int it = 0; it < 8; ++it) {
    int off4 = it * 32 + lane;       // 0..255
    int lr   = off4 >> 4;            // local row 0..15
    int row  = r0 + lr;
    if (row < N) {
      size_t i4 = base4 + off4;
      float sum   = scales[wave][lr];
      float scale = EPS_C / fmaxf(sqrtf(sum), 1e-12f);
      float4 xv  = xn4[i4];
      float4 nzv = nz4all[i4];
      float4 res;
      float sg;
      sg = (xv.x > 0.f) ? 1.f : ((xv.x < 0.f) ? -1.f : 0.f); res.x = fmaf(sg * scale, nzv.x, xv.x);
      sg = (xv.y > 0.f) ? 1.f : ((xv.y < 0.f) ? -1.f : 0.f); res.y = fmaf(sg * scale, nzv.y, xv.y);
      sg = (xv.z > 0.f) ? 1.f : ((xv.z < 0.f) ? -1.f : 0.f); res.z = fmaf(sg * scale, nzv.z, xv.z);
      sg = (xv.w > 0.f) ? 1.f : ((xv.w < 0.f) ? -1.f : 0.f); res.w = fmaf(sg * scale, nzv.w, xv.w);
      xc4[i4] = res;
      float4 o = ot4[i4];
      o.x += res.x * inv_layers; o.y += res.y * inv_layers;
      o.z += res.z * inv_layers; o.w += res.w * inv_layers;
      ot4[i4] = o;
      xn4[i4] = zero4;               // ready for next layer's atomic SpMM
    }
  }
}

extern "C" void kernel_launch(void* const* d_in, const int* in_sizes, int n_in,
                              void* d_out, int out_size, void* d_ws, size_t ws_size,
                              hipStream_t stream) {
  const float* Gu    = (const float*)d_in[0];
  const float* Gi    = (const float*)d_in[1];
  const float* noise = (const float*)d_in[2];
  const int*   ei    = (const int*)d_in[3];

  const int nu_elems = in_sizes[0];             // NUM_USERS * 64
  const int ni_elems = in_sizes[1];             // NUM_ITEMS * 64
  const int N        = (nu_elems + ni_elems) / 64;
  const int E        = in_sizes[3] / 2;         // directed edges
  const int total    = N * 64;
  const int L        = in_sizes[2] / total;     // layers (3)
  float* out = (float*)d_out;

  // workspace carve-out (256B aligned): dinv | norm | x_cur | x_next
  auto al = [](size_t x) { return (x + 255) & ~(size_t)255; };
  char* ws = (char*)d_ws;
  size_t o = 0;
  float* dinv   = (float*)(ws + o); o += al((size_t)N * 4);
  float* norm   = (float*)(ws + o); o += al((size_t)E * 4);
  float* x_cur  = (float*)(ws + o); o += al((size_t)total * 4);
  float* x_next = (float*)(ws + o); o += al((size_t)total * 4);
  (void)ws_size;

  simgcl_init<<<(total + 255) / 256, 256, 0, stream>>>(Gu, Gi, x_cur, x_next, out, dinv,
                                                       nu_elems, total, N);
  simgcl_deg<<<(E + 255) / 256, 256, 0, stream>>>(ei, dinv, E);
  simgcl_dinv<<<(N + 255) / 256, 256, 0, stream>>>(dinv, N);
  simgcl_norm<<<(E + 255) / 256, 256, 0, stream>>>(ei, dinv, norm, E);

  const float invL = 1.0f / (float)L;
  const unsigned spmm_blocks  = (unsigned)(((long long)E * 32 + 255) / 256);
  const unsigned noise_blocks = (unsigned)((N + 127) / 128);
  for (int k = 0; k < L; ++k) {
    simgcl_spmm<<<spmm_blocks, 256, 0, stream>>>(ei, norm, x_cur, x_next, E);
    simgcl_noise_wmma<<<noise_blocks, 256, 0, stream>>>(noise + (size_t)k * total,
                                                        x_next, x_cur, out, N, invL);
  }
}
